// CoverageMamba3_58841051955394
// MI455X (gfx1250) — compile-verified, matrix operand order checked by reading, not run on hardware
//
#include <hip/hip_runtime.h>
#include <hip/hip_bf16.h>
#include <cstdint>
#include <cstddef>

// ---------------------------------------------------------------------------
// CoverageMamba3 on MI455X (gfx1250, wave32, WMMA + async-LDS DMA).
//
//   prep     : weights f32 -> bf16 (A pre-transposed for scan B-operand)
//   in_proj  : h = x @ in_w^T + in_b                        (VALU, tiny K=8)
//   per layer:
//     ln_conv: xn = LN(h); xc = depthwise_conv(xn) + cb     (LDS-tiled)
//     gemm   : u = xc @ Bw^T                                (bf16 WMMA,
//              weight staged to LDS via global_load_async_to_lds_b128)
//     scan   : h_t = tanh(h_{t-1} @ A + u_t)  -> H          (persistent WMMA,
//              A resident in LDS, h double-buffered in LDS,
//              u_t double-buffered in LDS via async DMA, ASYNCcnt-pipelined)
//     gemm   : h = H @ Cw^T + D*xc + h(residual)            (bf16 WMMA, C-init)
//   head     : LN -> relu(o1) -> sigmoid(o2)*100            (VALU, tiny)
// ---------------------------------------------------------------------------

typedef __attribute__((ext_vector_type(16))) __bf16 v16bf;
typedef __attribute__((ext_vector_type(8)))  __bf16 v8bf;
typedef __attribute__((ext_vector_type(8)))  float  v8f;

#define B_   128
#define L_   1024
#define DM_  256
#define NROWS (B_ * L_)
#define WPITCH 264   // bf16 row pitch (256+8): 528B, 16B-aligned, bank-staggered
#define HPITCH 264
#define UPITCH 260   // f32 row pitch (256+4): 1040B, 16B-aligned, bank-staggered

#if defined(__HIP_DEVICE_COMPILE__) && defined(__gfx1250__) && \
    __has_builtin(__builtin_amdgcn_global_load_async_to_lds_b128)
#define HAS_ASYNC_LDS 1
#else
#define HAS_ASYNC_LDS 0
#endif

#if HAS_ASYNC_LDS
// Builtin signature (from hipcc diagnostic): typed int4 pointers,
// addrspace(1) global source, addrspace(3) LDS destination, imm offset, imm cpol.
typedef __attribute__((__vector_size__(4 * sizeof(int)))) int v4i_t;
typedef __attribute__((address_space(1))) v4i_t g_v4i;
typedef __attribute__((address_space(3))) v4i_t l_v4i;
// 16-byte global -> LDS DMA (no VGPR round trip), tracked by ASYNCcnt.
__device__ __forceinline__ void async_cp16(const void* g, void* l) {
  __builtin_amdgcn_global_load_async_to_lds_b128((g_v4i*)g, (l_v4i*)l, 0, 0);
}
__device__ __forceinline__ void wait_async0() {
#if __has_builtin(__builtin_amdgcn_s_wait_asynccnt)
  __builtin_amdgcn_s_wait_asynccnt(0);
#else
  asm volatile("s_wait_asynccnt 0" ::: "memory");
#endif
}
#endif

// ISA 7.12.2, 16-bit A-matrix 16x32 (wave32): lane m = lane&15 holds row M=m.
// Lanes 0-15: elems 0-7 = K kb..kb+7 (kb=0), elems 8-15 = K 16..23.
// Lanes 16-31: kb=8, elems 8-15 = K 24..31.  -> two 16B chunks per fragment.
__device__ __forceinline__ v16bf load_frag16(const __bf16* rowbase_k0, int kb) {
  const __bf16* p0 = rowbase_k0 + kb;
  v8bf lo = *(const v8bf*)(p0);
  v8bf hi = *(const v8bf*)(p0 + 16);
  v16bf a;
#pragma unroll
  for (int i = 0; i < 8; ++i) { a[i] = lo[i]; a[i + 8] = hi[i]; }
  return a;
}

// ---------------------------------------------------------------------------
// Weight prep: f32 -> bf16 once per launch.  A is stored transposed
// (At[d][s] = A[s][d]) so the scan can DMA it straight into its padded LDS
// B-operand layout; Bw/Cw stay row-major (B-frag [k][n] = W[n][k]).
// ---------------------------------------------------------------------------
__global__ void prep_weights_kernel(const float* __restrict__ A,
                                    const float* __restrict__ Bw,
                                    const float* __restrict__ Cw,
                                    __bf16* __restrict__ At_b,
                                    __bf16* __restrict__ Bw_b,
                                    __bf16* __restrict__ Cw_b) {
  const int layer = blockIdx.x / 3;
  const int which = blockIdx.x % 3;
  const int tid = threadIdx.x;
  const int base = layer * 65536;
  if (which == 0) {
    const float* src = A + base;
    __bf16* dst = At_b + base;
    for (int idx = tid; idx < 65536; idx += 256) {
      const int s = idx >> 8, d = idx & 255;
      dst[d * 256 + s] = (__bf16)src[idx];            // transpose on convert
    }
  } else {
    const float* src = (which == 1 ? Bw : Cw) + base;
    __bf16* dst = (which == 1 ? Bw_b : Cw_b) + base;
    for (int idx = tid; idx < 65536; idx += 256) dst[idx] = (__bf16)src[idx];
  }
}

// ---------------------------------------------------------------------------
// Input projection: out[row, d] = sum_k x[row,k] * w[d,k] + b[d]   (K = 8)
// ---------------------------------------------------------------------------
__global__ void in_proj_kernel(const float* __restrict__ x,
                               const float* __restrict__ w,
                               const float* __restrict__ bias,
                               float* __restrict__ out) {
  const int row = blockIdx.x;
  const int d = threadIdx.x;
  const float* xr = x + (size_t)row * 8;   // block-uniform -> scalar loads
  const float* wr = w + d * 8;
  float acc = bias[d];
#pragma unroll
  for (int k = 0; k < 8; ++k) acc += xr[k] * wr[k];
  out[(size_t)row * DM_ + d] = acc;
}

// ---------------------------------------------------------------------------
// LayerNorm + depthwise conv1d (DC=4, left pad 2).  One block = one batch b,
// 16 timesteps; 19 normalized timesteps (halo) staged in LDS.
// ---------------------------------------------------------------------------
__global__ void ln_conv_kernel(const float* __restrict__ hin,
                               float* __restrict__ xcf, __bf16* __restrict__ xcb,
                               const float* __restrict__ g, const float* __restrict__ bt,
                               const float* __restrict__ cw,   // [256,4]
                               const float* __restrict__ cb) {
  __shared__ float xn[19 * 264];
  const int tid = threadIdx.x;                 // 256 threads = 8 waves
  const int wave = tid >> 5, lane = tid & 31;
  const int b = blockIdx.x >> 6;               // 64 chunks of 16 steps
  const int t0 = (blockIdx.x & 63) * 16;

  for (int ts = wave; ts < 19; ts += 8) {
    const int t = t0 - 2 + ts;
    if (t < 0 || t >= L_) {
#pragma unroll
      for (int j = 0; j < 8; ++j) xn[ts * 264 + lane + 32 * j] = 0.f;
    } else {
      const float* row = hin + (size_t)(b * L_ + t) * DM_;
      float vals[8]; float s = 0.f;
#pragma unroll
      for (int j = 0; j < 8; ++j) { vals[j] = row[lane + 32 * j]; s += vals[j]; }
#pragma unroll
      for (int m = 16; m >= 1; m >>= 1) s += __shfl_xor(s, m, 32);
      const float mean = s * (1.f / 256.f);
      float v = 0.f;
#pragma unroll
      for (int j = 0; j < 8; ++j) { float d = vals[j] - mean; v += d * d; }
#pragma unroll
      for (int m = 16; m >= 1; m >>= 1) v += __shfl_xor(v, m, 32);
      const float rs = rsqrtf(v * (1.f / 256.f) + 1e-5f);
#pragma unroll
      for (int j = 0; j < 8; ++j) {
        const int d = lane + 32 * j;
        xn[ts * 264 + d] = (vals[j] - mean) * rs * g[d] + bt[d];
      }
    }
  }
  __syncthreads();

  const int d = tid;
  const float w0 = cw[d * 4 + 0], w1 = cw[d * 4 + 1];
  const float w2 = cw[d * 4 + 2], w3 = cw[d * 4 + 3];
  const float bias = cb[d];
  for (int tt = 0; tt < 16; ++tt) {
    const float acc = bias
        + xn[(tt + 0) * 264 + d] * w0 + xn[(tt + 1) * 264 + d] * w1
        + xn[(tt + 2) * 264 + d] * w2 + xn[(tt + 3) * 264 + d] * w3;
    const size_t idx = (size_t)(b * L_ + t0 + tt) * DM_ + d;
    xcf[idx] = acc;
    xcb[idx] = (__bf16)acc;
  }
}

// ---------------------------------------------------------------------------
// 256-wide bf16 WMMA GEMM:  out = Abf @ Wb^T [+ D*xc + res].
// Wb [256,256] bf16 row-major, DMA'd into padded LDS via async-to-LDS b128.
// Block = 8 waves x 16 rows = 128 rows; wave preloads full-K A fragments and
// sweeps 16 N-tiles.
// ---------------------------------------------------------------------------
__global__ void gemm256_kernel(const __bf16* __restrict__ Abf,
                               const __bf16* __restrict__ Wb,
                               float* __restrict__ out,
                               int mode,
                               const float* __restrict__ Dv,
                               const float* __restrict__ xcf,
                               const float* __restrict__ res) {
  extern __shared__ __align__(16) char smem[];
  __bf16* wl = (__bf16*)smem;                       // [256][WPITCH]
  const int tid = threadIdx.x;                      // 256 threads
#if HAS_ASYNC_LDS
  for (int chunk = tid; chunk < 256 * 32; chunk += 256) {   // 16B chunks
    const int r = chunk >> 5, c8 = (chunk & 31) * 8;
    async_cp16(Wb + r * 256 + c8, wl + r * WPITCH + c8);
  }
  wait_async0();
#else
  for (int idx = tid; idx < 256 * 256; idx += 256)
    wl[(idx >> 8) * WPITCH + (idx & 255)] = Wb[idx];
#endif
  __syncthreads();

  const int wave = tid >> 5, lane = tid & 31;
  const int m16 = lane & 15;
  const int kb  = ((lane >> 4) & 1) * 8;
  const int mb  = (lane >> 4) * 8;
  const int row0 = blockIdx.x * 128 + wave * 16;

  v16bf afr[8];
  const __bf16* arow = Abf + (size_t)(row0 + m16) * DM_;
#pragma unroll
  for (int ks = 0; ks < 8; ++ks) afr[ks] = load_frag16(arow + ks * 32, kb);

  for (int nt = 0; nt < 16; ++nt) {
    const int col = nt * 16 + m16;
    v8f c;
    if (mode == 0) {
#pragma unroll
      for (int r = 0; r < 8; ++r) c[r] = 0.f;
    } else {
      const float dval = Dv[col];
#pragma unroll
      for (int r = 0; r < 8; ++r) {
        const size_t idx = (size_t)(row0 + mb + r) * DM_ + col;
        c[r] = res[idx] + dval * xcf[idx];
      }
    }
#pragma unroll
    for (int ks = 0; ks < 8; ++ks) {
      v16bf bf = load_frag16(wl + (size_t)(nt * 16 + m16) * WPITCH + ks * 32, kb);
      c = __builtin_amdgcn_wmma_f32_16x16x32_bf16(false, afr[ks], false, bf,
                                                  (short)0, c, false, false);
    }
#pragma unroll
    for (int r = 0; r < 8; ++r)
      out[(size_t)(row0 + mb + r) * DM_ + col] = c[r];
  }
}

// ---------------------------------------------------------------------------
// Persistent recurrent scan: h_t = tanh(h_{t-1} @ A + u_t), store h_t (bf16).
// 8 blocks x 16 batch rows (independent recurrences).  A^T resident in LDS
// (B-operand, DMA-staged); h double-buffered in LDS; u_t double-buffered in
// LDS by async DMA one step ahead (ASYNCcnt-pipelined).  16 waves each own
// one 16-col N-tile -> 8 WMMAs/wave/step, 1 barrier/step.
// ---------------------------------------------------------------------------
__global__ void scan_kernel(const float* __restrict__ u,    // [NROWS,256], row=b*L+t
                            const __bf16* __restrict__ Atb, // [256,256] At[d][s]
                            __bf16* __restrict__ H) {       // [NROWS,256]
  extern __shared__ __align__(16) char smem[];
  __bf16* al = (__bf16*)smem;                               // [256][WPITCH]
  __bf16* h0 = (__bf16*)(smem + 256 * WPITCH * sizeof(__bf16));
  __bf16* h1 = h0 + 16 * HPITCH;
  float* ul0 = (float*)(smem + 256 * WPITCH * sizeof(__bf16)
                             + 2 * 16 * HPITCH * sizeof(__bf16));
  float* ul1 = ul0 + 16 * UPITCH;

  const int tid = threadIdx.x;                              // 512 threads = 16 waves
  const int b0 = blockIdx.x * 16;

#if HAS_ASYNC_LDS
  for (int chunk = tid; chunk < 256 * 32; chunk += 512) {   // A^T -> LDS (DMA)
    const int r = chunk >> 5, c8 = (chunk & 31) * 8;
    async_cp16(Atb + r * 256 + c8, al + r * WPITCH + c8);
  }
#else
  for (int idx = tid; idx < 256 * 256; idx += 512)
    al[(idx >> 8) * WPITCH + (idx & 255)] = Atb[idx];
#endif
  for (int idx = tid; idx < 16 * HPITCH; idx += 512) h0[idx] = (__bf16)0.f;
#if HAS_ASYNC_LDS
  { // stage u_0
    for (int chunk = tid; chunk < 16 * 64; chunk += 512) {  // 16B = 4 f32
      const int r = chunk >> 6, c4 = (chunk & 63) * 4;
      async_cp16(u + ((size_t)(b0 + r) * L_ + 0) * DM_ + c4, ul0 + r * UPITCH + c4);
    }
    wait_async0();
  }
#endif
  __syncthreads();

  const int wave = tid >> 5, lane = tid & 31;
  const int nt  = wave;                                     // one N-tile per wave
  const int m16 = lane & 15;
  const int kb  = ((lane >> 4) & 1) * 8;
  const int mb  = (lane >> 4) * 8;
  const int col = nt * 16 + m16;

  __bf16* hcur = h0;  __bf16* hnext = h1;
  float*  ucur = ul0; float*  unext = ul1;

  for (int t = 0; t < L_; ++t) {
#if HAS_ASYNC_LDS
    if (t + 1 < L_) {                                       // prefetch u_{t+1} -> LDS
      for (int chunk = tid; chunk < 16 * 64; chunk += 512) {
        const int r = chunk >> 6, c4 = (chunk & 63) * 4;
        async_cp16(u + ((size_t)(b0 + r) * L_ + (t + 1)) * DM_ + c4,
                   unext + r * UPITCH + c4);
      }
    }
    v8f c;
#pragma unroll
    for (int r = 0; r < 8; ++r) c[r] = ucur[(mb + r) * UPITCH + col];
#else
    v8f c;
#pragma unroll
    for (int r = 0; r < 8; ++r)
      c[r] = u[((size_t)(b0 + mb + r) * L_ + t) * DM_ + col];
    if (t + 1 < L_) {                                       // global_prefetch_b8
#pragma unroll
      for (int r = 0; r < 8; ++r)
        __builtin_prefetch(&u[((size_t)(b0 + mb + r) * L_ + (t + 1)) * DM_ + col], 0, 1);
    }
#endif
#pragma unroll
    for (int ks = 0; ks < 8; ++ks) {
      v16bf af = load_frag16(hcur + m16 * HPITCH + ks * 32, kb);
      v16bf bf = load_frag16(al + (size_t)col * WPITCH + ks * 32, kb);
      c = __builtin_amdgcn_wmma_f32_16x16x32_bf16(false, af, false, bf,
                                                  (short)0, c, false, false);
    }
#pragma unroll
    for (int r = 0; r < 8; ++r) {
      const __bf16 bv = (__bf16)tanhf(c[r]);
      hnext[(mb + r) * HPITCH + col] = bv;
      H[((size_t)(b0 + mb + r) * L_ + t) * DM_ + col] = bv;
    }
#if HAS_ASYNC_LDS
    wait_async0();
#endif
    __syncthreads();
    { __bf16* tb = hcur; hcur = hnext; hnext = tb; }
    { float*  tu = ucur; ucur = unext; unext = tu; }
  }
}

// ---------------------------------------------------------------------------
// Output head: LN(h[:, L-1, :]) -> relu(@o1_w^T+o1_b) -> sigmoid(@o2_w^T+o2_b)*100
// ---------------------------------------------------------------------------
__global__ void head_kernel(const float* __restrict__ h,
                            const float* __restrict__ g, const float* __restrict__ bt,
                            const float* __restrict__ w1, const float* __restrict__ b1,
                            const float* __restrict__ w2, const float* __restrict__ b2,
                            float* __restrict__ out) {
  __shared__ float zn[256];
  __shared__ float red[128];
  const int b = blockIdx.x, tid = threadIdx.x;              // 128 threads
  const float* row = h + ((size_t)b * L_ + (L_ - 1)) * DM_;
  const float a0 = row[tid], a1 = row[tid + 128];
  red[tid] = a0 + a1; __syncthreads();
  for (int s = 64; s >= 1; s >>= 1) { if (tid < s) red[tid] += red[tid + s]; __syncthreads(); }
  const float mean = red[0] * (1.f / 256.f); __syncthreads();
  const float d0 = a0 - mean, d1 = a1 - mean;
  red[tid] = d0 * d0 + d1 * d1; __syncthreads();
  for (int s = 64; s >= 1; s >>= 1) { if (tid < s) red[tid] += red[tid + s]; __syncthreads(); }
  const float rs = rsqrtf(red[0] * (1.f / 256.f) + 1e-5f); __syncthreads();
  zn[tid] = d0 * rs * g[tid] + bt[tid];
  zn[tid + 128] = d1 * rs * g[tid + 128] + bt[tid + 128];
  __syncthreads();
  float acc = b1[tid];
  const float* wr = w1 + tid * 256;
  for (int k = 0; k < 256; ++k) acc += zn[k] * wr[k];
  acc = fmaxf(acc, 0.f);
  red[tid] = acc * w2[tid]; __syncthreads();
  for (int s = 64; s >= 1; s >>= 1) { if (tid < s) red[tid] += red[tid + s]; __syncthreads(); }
  if (tid == 0) out[b] = 100.f / (1.f + expf(-(red[0] + b2[0])));
}

// ---------------------------------------------------------------------------
extern "C" void kernel_launch(void* const* d_in, const int* in_sizes, int n_in,
                              void* d_out, int out_size, void* d_ws, size_t ws_size,
                              hipStream_t stream) {
  const float* x    = (const float*)d_in[0];
  const float* in_w = (const float*)d_in[1];
  const float* in_b = (const float*)d_in[2];
  const float* A    = (const float*)d_in[3];
  const float* Bw   = (const float*)d_in[4];
  const float* Cw   = (const float*)d_in[5];
  const float* Dv   = (const float*)d_in[6];
  const float* cw   = (const float*)d_in[7];
  const float* cb   = (const float*)d_in[8];
  const float* lng  = (const float*)d_in[9];
  const float* lnb  = (const float*)d_in[10];
  const float* lnog = (const float*)d_in[11];
  const float* lnob = (const float*)d_in[12];
  const float* o1w  = (const float*)d_in[13];
  const float* o1b  = (const float*)d_in[14];
  const float* o2w  = (const float*)d_in[15];
  const float* o2b  = (const float*)d_in[16];
  float* out = (float*)d_out;

  char* ws = (char*)d_ws;
  const size_t F32BUF  = (size_t)NROWS * DM_ * sizeof(float);   // 128 MiB
  const size_t BF16BUF = (size_t)NROWS * DM_ * sizeof(__bf16);  // 64 MiB
  const size_t WBUF    = (size_t)4 * 65536 * sizeof(__bf16);    // 512 KiB / weight set
  float*  buf_h = (float*)(ws);
  float*  xc_f  = (float*)(ws + F32BUF);
  float*  u_f   = (float*)(ws + 2 * F32BUF);
  __bf16* xc_b  = (__bf16*)(ws + 3 * F32BUF);
  __bf16* H_b   = (__bf16*)(ws + 3 * F32BUF + BF16BUF);
  __bf16* At_b  = (__bf16*)(ws + 3 * F32BUF + 2 * BF16BUF);
  __bf16* Bw_b  = (__bf16*)(ws + 3 * F32BUF + 2 * BF16BUF + WBUF);
  __bf16* Cw_b  = (__bf16*)(ws + 3 * F32BUF + 2 * BF16BUF + 2 * WBUF);

  const size_t GEMM_LDS = (size_t)256 * WPITCH * sizeof(__bf16);           // 132 KiB
  const size_t SCAN_LDS = GEMM_LDS + (size_t)2 * 16 * HPITCH * sizeof(__bf16)
                                   + (size_t)2 * 16 * UPITCH * sizeof(float);

  prep_weights_kernel<<<12, 256, 0, stream>>>(A, Bw, Cw, At_b, Bw_b, Cw_b);
  in_proj_kernel<<<NROWS, 256, 0, stream>>>(x, in_w, in_b, buf_h);
  for (int i = 0; i < 4; ++i) {
    ln_conv_kernel<<<B_ * (L_ / 16), 256, 0, stream>>>(
        buf_h, xc_f, xc_b, lng + i * 256, lnb + i * 256, cw + i * 1024, cb + i * 256);
    gemm256_kernel<<<NROWS / 128, 256, GEMM_LDS, stream>>>(
        xc_b, Bw_b + i * 65536, u_f, 0, nullptr, nullptr, nullptr);
    scan_kernel<<<8, 512, SCAN_LDS, stream>>>(u_f, At_b + i * 65536, H_b);
    gemm256_kernel<<<NROWS / 128, 256, GEMM_LDS, stream>>>(
        H_b, Cw_b + i * 65536, buf_h, 1, Dv + i * 256, xc_f, buf_h);
  }
  head_kernel<<<B_, 128, 0, stream>>>(buf_h, lnog, lnob, o1w, o1b, o2w, o2b, out);
}
